// LinearBinary_48180943127010
// MI455X (gfx1250) — compile-verified, matrix operand order checked by reading, not run on hardware
//
#include <hip/hip_runtime.h>

typedef _Float16 half_t;
typedef _Float16 v16h __attribute__((ext_vector_type(16)));
typedef _Float16 v8h  __attribute__((ext_vector_type(8)));
typedef float    v8f  __attribute__((ext_vector_type(8)));
typedef int      v4i  __attribute__((ext_vector_type(4)));

typedef __attribute__((address_space(1))) v4i* gptr4_t;  // global
typedef __attribute__((address_space(3))) v4i* sptr4_t;  // LDS

#ifndef __has_builtin
#define __has_builtin(x) 0
#endif

#if __has_builtin(__builtin_amdgcn_global_load_async_to_lds_b128)
#define USE_ASYNC_COPY 1
#else
#define USE_ASYNC_COPY 0
#endif

#define GEMM_M 8192
#define GEMM_N 4096
#define GEMM_K 4096

#define TILE_M 128
#define TILE_N 128
#define TILE_K 64
#define LDS_K  (TILE_K + 8)   // 72 halfs = 144 B row stride: 16B-aligned, bank-conflict-free
#define THREADS 256

// ---------------- conversion kernels ----------------

__global__ void LB_cvt_x_f16(const float* __restrict__ src, half_t* __restrict__ dst, int n8) {
    int i = blockIdx.x * blockDim.x + threadIdx.x;
    if (i >= n8) return;
    const float4* p = reinterpret_cast<const float4*>(src) + (size_t)i * 2;
    float4 a = p[0];
    float4 b = p[1];
    v8h o;
    o[0] = (half_t)a.x; o[1] = (half_t)a.y; o[2] = (half_t)a.z; o[3] = (half_t)a.w;
    o[4] = (half_t)b.x; o[5] = (half_t)b.y; o[6] = (half_t)b.z; o[7] = (half_t)b.w;
    reinterpret_cast<v8h*>(dst)[i] = o;
}

__device__ __forceinline__ half_t binarize1(float v) {
    // forward of STE: clamp(round-half-even(v), 0, 1); exact in f16
    float r = __builtin_rintf(v);
    r = fminf(fmaxf(r, 0.0f), 1.0f);
    return (half_t)r;
}

__global__ void LB_cvt_w_bin_f16(const float* __restrict__ src, half_t* __restrict__ dst, int n8) {
    int i = blockIdx.x * blockDim.x + threadIdx.x;
    if (i >= n8) return;
    const float4* p = reinterpret_cast<const float4*>(src) + (size_t)i * 2;
    float4 a = p[0];
    float4 b = p[1];
    v8h o;
    o[0] = binarize1(a.x); o[1] = binarize1(a.y); o[2] = binarize1(a.z); o[3] = binarize1(a.w);
    o[4] = binarize1(b.x); o[5] = binarize1(b.y); o[6] = binarize1(b.z); o[7] = binarize1(b.w);
    reinterpret_cast<v8h*>(dst)[i] = o;
}

// ---------------- WMMA GEMM:  C[M,N] = A[M,K] * B[N,K]^T + bias ----------------
// NT GEMM: both A and B are K-contiguous, so A-frag and B-frag use the same
// lane layout (ISA 16-bit A-matrix 16x32: lane = row, lanes>=16 start at K=8;
// each lane holds K-chunks [kb..kb+7] and [kb+16..kb+23], kb = (lane>>4)*8).

__device__ __forceinline__ v16h load_frag_lds(const half_t* tile, int row, int kk, int lane) {
    const int kb = (lane >> 4) * 8;
    const half_t* p = tile + row * LDS_K + kk + kb;
    v8h lo = *reinterpret_cast<const v8h*>(p);
    v8h hi = *reinterpret_cast<const v8h*>(p + 16);
    return __builtin_shufflevector(lo, hi, 0, 1, 2, 3, 4, 5, 6, 7,
                                            8, 9, 10, 11, 12, 13, 14, 15);
}

__launch_bounds__(THREADS)
__global__ void LB_wmma_gemm(const half_t* __restrict__ A,   // [8192,4096] f16 x
                             const half_t* __restrict__ B,   // [4096,4096] f16 binarized w
                             const float*  __restrict__ bias, // [4096]
                             float* __restrict__ C)           // [8192,4096]
{
    __shared__ half_t Asmem[2][TILE_M * LDS_K];
    __shared__ half_t Bsmem[2][TILE_N * LDS_K];

    const int tid  = threadIdx.x;
    const int lane = tid & 31;
    const int wave = tid >> 5;
    const int wm   = wave & 3;   // 4 waves along M: 32 rows each
    const int wn   = wave >> 2;  // 2 waves along N: 64 cols each

    const int m0 = blockIdx.y * TILE_M;
    const int n0 = blockIdx.x * TILE_N;

    v8f zero8 = {};
    v8f acc[2][4];
#pragma unroll
    for (int mi = 0; mi < 2; ++mi)
#pragma unroll
        for (int ni = 0; ni < 4; ++ni) acc[mi][ni] = zero8;

    // stage one 128x64 A tile + 128x64 B tile into LDS buffer `buf`
    auto stage = [&](int buf, int kbase) {
#pragma unroll
        for (int i = 0; i < 4; ++i) {
            int c   = tid + i * THREADS;   // 0..1023 chunk id (8 halfs per chunk)
            int row = c >> 3;
            int col = (c & 7) * 8;
            const half_t* ga = A + (size_t)(m0 + row) * GEMM_K + kbase + col;
            const half_t* gb = B + (size_t)(n0 + row) * GEMM_K + kbase + col;
            half_t* la = &Asmem[buf][row * LDS_K + col];
            half_t* lb = &Bsmem[buf][row * LDS_K + col];
#if USE_ASYNC_COPY
            __builtin_amdgcn_global_load_async_to_lds_b128(
                (gptr4_t)ga, (sptr4_t)la, 0, 0);
            __builtin_amdgcn_global_load_async_to_lds_b128(
                (gptr4_t)gb, (sptr4_t)lb, 0, 0);
#else
            *reinterpret_cast<v8h*>(la) = *reinterpret_cast<const v8h*>(ga);
            *reinterpret_cast<v8h*>(lb) = *reinterpret_cast<const v8h*>(gb);
#endif
        }
    };

    auto commit = [&]() {
#if USE_ASYNC_COPY
#if __has_builtin(__builtin_amdgcn_s_wait_asynccnt)
        __builtin_amdgcn_s_wait_asynccnt(0);
#else
        asm volatile("s_wait_asynccnt 0" ::: "memory");
#endif
#endif
        __syncthreads();
    };

    stage(0, 0);
    commit();

    const int KSTAGES = GEMM_K / TILE_K;   // 64
    for (int s = 0; s < KSTAGES; ++s) {
        const int buf = s & 1;
        if (s + 1 < KSTAGES) stage(buf ^ 1, (s + 1) * TILE_K);

        const half_t* At = Asmem[buf];
        const half_t* Bt = Bsmem[buf];
        const int r = lane & 15;

#pragma unroll
        for (int kk = 0; kk < TILE_K; kk += 32) {
            v16h af[2], bf[4];
#pragma unroll
            for (int mi = 0; mi < 2; ++mi)
                af[mi] = load_frag_lds(At, wm * 32 + mi * 16 + r, kk, lane);
#pragma unroll
            for (int ni = 0; ni < 4; ++ni)
                bf[ni] = load_frag_lds(Bt, wn * 64 + ni * 16 + r, kk, lane);
#pragma unroll
            for (int mi = 0; mi < 2; ++mi)
#pragma unroll
                for (int ni = 0; ni < 4; ++ni)
                    acc[mi][ni] = __builtin_amdgcn_wmma_f32_16x16x32_f16(
                        false, af[mi], false, bf[ni],
                        (short)0, acc[mi][ni], false, false);
        }
        commit();
    }

    // Epilogue: C layout — VGPR r holds M = r + 8*(lane>=16); lane%16 = N col
#pragma unroll
    for (int mi = 0; mi < 2; ++mi) {
        const int row0 = m0 + wm * 32 + mi * 16 + (lane >> 4) * 8;
#pragma unroll
        for (int ni = 0; ni < 4; ++ni) {
            const int col = n0 + wn * 64 + ni * 16 + (lane & 15);
            const float bv = bias[col];
#pragma unroll
            for (int rr = 0; rr < 8; ++rr)
                C[(size_t)(row0 + rr) * GEMM_N + col] = acc[mi][ni][rr] + bv;
        }
    }
}

// ---------------- launcher ----------------

extern "C" void kernel_launch(void* const* d_in, const int* in_sizes, int n_in,
                              void* d_out, int out_size, void* d_ws, size_t ws_size,
                              hipStream_t stream) {
    const float* x    = (const float*)d_in[0];
    const float* w    = (const float*)d_in[1];
    const float* bias = (const float*)d_in[2];
    float* out = (float*)d_out;

    half_t* xh = (half_t*)d_ws;                             // 8192*4096*2 = 64 MiB
    half_t* wh = xh + (size_t)GEMM_M * GEMM_K;              // 4096*4096*2 = 32 MiB

    {
        int n8 = GEMM_M * GEMM_K / 8;
        LB_cvt_x_f16<<<(n8 + 255) / 256, 256, 0, stream>>>(x, xh, n8);
    }
    {
        int n8 = GEMM_N * GEMM_K / 8;
        LB_cvt_w_bin_f16<<<(n8 + 255) / 256, 256, 0, stream>>>(w, wh, n8);
    }
    dim3 grid(GEMM_N / TILE_N, GEMM_M / TILE_M);  // (32, 64)
    LB_wmma_gemm<<<grid, THREADS, 0, stream>>>(xh, wh, bias, out);
}
